// CausalAttn_63806034149965
// MI455X (gfx1250) — matrix operand order repaired and checked by reading, not compiled, not run on hardware
//
#include <hip/hip_runtime.h>
#include <hip/hip_bf16.h>
#include <math.h>

typedef __bf16 bf16_t;
typedef __attribute__((ext_vector_type(16))) __bf16 v16bf;
typedef __attribute__((ext_vector_type(8)))  __bf16 v8bf;
typedef __attribute__((ext_vector_type(8)))  float  v8f;

union V16U { v16bf v; v8bf h[2]; };

#define WMMA_BF16(a, b, c) \
  __builtin_amdgcn_wmma_f32_16x16x32_bf16(false, (a), false, (b), (short)0, (c), false, false)

// ---------------------------------------------------------------------------
// WMMA operand loaders (wave32 register layouts per CDNA5 ISA 7.12.2)
// A (16x32 bf16): lane L -> m = L%16 ; dwords0-3 = K kb..kb+7, dwords4-7 =
// K kb+16..kb+23 with kb = (L/16)*8.  Two contiguous 16B loads.
// ---------------------------------------------------------------------------
__device__ __forceinline__ v16bf load_a(const bf16_t* p, int ldm) {
  int lane = threadIdx.x & 31;
  int m  = lane & 15;
  int kb = (lane >> 4) * 8;
  V16U u;
  u.h[0] = *(const v8bf*)(p + (size_t)m * ldm + kb);
  u.h[1] = *(const v8bf*)(p + (size_t)m * ldm + kb + 16);
  return u.v;
}

// B (32x16 bf16) with storage as BT[n][k] (N-major): lane L -> n = L%16,
// K kb..kb+7 and kb+16..kb+23 with kb = (L/16)*8 (2x16B loads).
__device__ __forceinline__ v16bf load_bt(const bf16_t* p, int ldn) {
  int lane = threadIdx.x & 31;
  int n  = lane & 15;
  int kb = (lane >> 4) * 8;
  V16U u;
  u.h[0] = *(const v8bf*)(p + (size_t)n * ldn + kb);
  u.h[1] = *(const v8bf*)(p + (size_t)n * ldn + kb + 16);
  return u.v;
}

// ---------------------------------------------------------------------------
// Prep kernels
// ---------------------------------------------------------------------------
__global__ void cvt_bf16_kernel(const float* __restrict__ src,
                                bf16_t* __restrict__ dst, int n) {
  int i = blockIdx.x * blockDim.x + threadIdx.x;
  if (i < n) dst[i] = (bf16_t)src[i];
}

// src (R,C) f32 -> dst (C,R) bf16  (transpose + convert)
__global__ void cvt_transpose_kernel(const float* __restrict__ src,
                                     bf16_t* __restrict__ dst, int R, int C) {
  int i = blockIdx.x * blockDim.x + threadIdx.x;
  if (i < R * C) {
    int r = i % R;
    int c = i / R;
    dst[i] = (bf16_t)src[(size_t)r * C + c];
  }
}

__global__ void compute_len_kernel(const unsigned char* __restrict__ mask,
                                   int* __restrict__ lens) {
  __shared__ int red[256];
  int b = blockIdx.x;
  int s = 0;
  for (int t = threadIdx.x; t < 2048; t += 256) s += mask[b * 2048 + t] ? 1 : 0;
  red[threadIdx.x] = s;
  __syncthreads();
  for (int o = 128; o > 0; o >>= 1) {
    if (threadIdx.x < (unsigned)o) red[threadIdx.x] += red[threadIdx.x + o];
    __syncthreads();
  }
  if (threadIdx.x == 0) lens[b] = red[0];
}

// ---------------------------------------------------------------------------
// Kernel 1: QKV projection.  C(8192,3072) = Xb(8192,1024) x W + bias.
// Each wave computes a 64x64 tile: 4 A ops x 4 B ops -> 16 WMMAs / k-step
// (~32 FLOP per L2 byte).  Q,K written as (b,h,T,64) bf16; V transposed
// as (b,h,64,T) bf16.
// ---------------------------------------------------------------------------
__global__ void __launch_bounds__(256)
qkv_gemm_kernel(const bf16_t* __restrict__ Xb, const bf16_t* __restrict__ WT,
                const float* __restrict__ bias, bf16_t* __restrict__ Qb,
                bf16_t* __restrict__ Kb, bf16_t* __restrict__ VTb) {
  const int K = 1024, T = 2048, NT = 48;  // 3072/64 n-tiles
  int wg = blockIdx.x * 8 + (threadIdx.x >> 5);
  int mt = wg / NT, nt = wg % NT;
  int m0 = mt * 64, n0 = nt * 64;
  int lane = threadIdx.x & 31;

  const v8f vz = {0.f, 0.f, 0.f, 0.f, 0.f, 0.f, 0.f, 0.f};
  v8f acc[4][4];
#pragma unroll
  for (int i = 0; i < 4; ++i)
#pragma unroll
    for (int j = 0; j < 4; ++j) acc[i][j] = vz;

  for (int k0 = 0; k0 < K; k0 += 32) {
    v16bf a[4];
#pragma unroll
    for (int i = 0; i < 4; ++i)
      a[i] = load_a(Xb + (size_t)(m0 + i * 16) * K + k0, K);
#pragma unroll
    for (int j = 0; j < 4; ++j) {
      v16bf b = load_bt(WT + (size_t)(n0 + j * 16) * K + k0, K);
#pragma unroll
      for (int i = 0; i < 4; ++i) acc[i][j] = WMMA_BF16(a[i], b, acc[i][j]);
    }
  }

  // Epilogue.  C layout: lane -> n = lane%16, rows m = (lane/16)*8 + r.
  int nloc = lane & 15;
  int moff = (lane >> 4) * 8;
#pragma unroll
  for (int i = 0; i < 4; ++i) {
#pragma unroll
    for (int j = 0; j < 4; ++j) {
      int col = n0 + j * 16 + nloc;
      float bv  = bias[col];
      int sec = col >> 10;          // 0=Q 1=K 2=V
      int c   = col & 1023;
      int h = c >> 6, d = c & 63;
      int mbase = m0 + i * 16 + moff;      // 8 consecutive rows, same batch
      int bidx  = mbase / T;
      int t0    = mbase % T;
      int bh    = bidx * 16 + h;
      if (sec == 2) {
        v8bf pk;
#pragma unroll
        for (int r = 0; r < 8; ++r) pk[r] = (bf16_t)(acc[i][j][r] + bv);
        *(v8bf*)(VTb + ((size_t)bh * 64 + d) * T + t0) = pk;  // (b,h,64,T)
      } else {
        bf16_t* dst = (sec == 0 ? Qb : Kb) + ((size_t)bh * T + t0) * 64 + d;
#pragma unroll
        for (int r = 0; r < 8; ++r) dst[(size_t)r * 64] = (bf16_t)(acc[i][j][r] + bv);
      }
    }
  }
}

// ---------------------------------------------------------------------------
// Kernel 2: flash attention.  One block = one (b,h), 128 consecutive q-rows
// (8 waves x 16-row q-tiles).  Each 32-key K/V block is staged into LDS once
// per block and shared by all 8 waves (8x less K/V traffic than per-wave
// loading).  Output written bf16 into (B,T,H) for the final projection.
// ---------------------------------------------------------------------------
__global__ void __launch_bounds__(256)
attn_kernel(const bf16_t* __restrict__ Qb, const bf16_t* __restrict__ Kb,
            const bf16_t* __restrict__ VTb, const int* __restrict__ lens,
            bf16_t* __restrict__ Ob) {
  const int T = 2048, HD = 64;
  __shared__ __align__(16) bf16_t ldsK[32 * 64];       // (key,d)  4KB
  __shared__ __align__(16) bf16_t ldsV[64 * 32];       // (d,key)  4KB
  __shared__ __align__(16) bf16_t pstage[8][16 * 32];  // per-wave P 8KB

  int tid  = threadIdx.x;
  int wave = tid >> 5;
  int lane = tid & 31;
  int bh = blockIdx.x >> 4;          // 64 (b,h) pairs
  int qb = blockIdx.x & 15;          // 16 blocks of 128 q-rows
  int b  = bh >> 4;
  int h  = bh & 15;
  int q0 = qb * 128 + wave * 16;     // this wave's 16-row q-tile

  const bf16_t* Qp = Qb  + (size_t)bh * T * HD;
  const bf16_t* Kp = Kb  + (size_t)bh * T * HD;
  const bf16_t* Vp = VTb + (size_t)bh * HD * T;
  int len = lens[b];

  v16bf aq0 = load_a(Qp + (size_t)q0 * HD, HD);       // d 0..31
  v16bf aq1 = load_a(Qp + (size_t)q0 * HD + 32, HD);  // d 32..63

  const v8f vz = {0.f, 0.f, 0.f, 0.f, 0.f, 0.f, 0.f, 0.f};
  v8f accO[4];
  float mrow[8], lrow[8];
#pragma unroll
  for (int j = 0; j < 4; ++j) accO[j] = vz;
#pragma unroll
  for (int r = 0; r < 8; ++r) { mrow[r] = -3.0e38f; lrow[r] = 0.f; }

  int nloc = lane & 15;
  int moff = (lane >> 4) * 8;
  int kend_w = min(q0 + 16, len);              // this wave's key bound
  int kend_b = min(qb * 128 + 128, len);       // block-wide key bound

  // K staging: thread -> (row = tid/8, 16B chunk = tid%8)   (32x64 bf16)
  int krow = tid >> 3, kch = (tid & 7) * 8;
  // V staging: thread -> (d = tid/4, 16B chunk = tid%4)     (64x32 bf16)
  int vd = tid >> 2, vch = (tid & 3) * 8;

  for (int k0 = 0; k0 < kend_b; k0 += 32) {
    __syncthreads();   // previous iteration's LDS reads complete
    *(v8bf*)(ldsK + krow * 64 + kch) =
        *(const v8bf*)(Kp + (size_t)(k0 + krow) * HD + kch);
    *(v8bf*)(ldsV + vd * 32 + vch) =
        *(const v8bf*)(Vp + (size_t)vd * T + k0 + vch);
    __syncthreads();   // K/V block visible to all waves

    if (k0 < kend_w) {
      // ---- scores: 16x32 tile = two 16x16 WMMAs x 2 K-chunks ----
      v8f s0 = vz, s1 = vz;
      {
        v16bf bk;
        bk = load_bt(ldsK + 0 * 16 * 64 + 0, 64);  s0 = WMMA_BF16(aq0, bk, s0);
        bk = load_bt(ldsK + 0 * 16 * 64 + 32, 64); s0 = WMMA_BF16(aq1, bk, s0);
        bk = load_bt(ldsK + 1 * 16 * 64 + 0, 64);  s1 = WMMA_BF16(aq0, bk, s1);
        bk = load_bt(ldsK + 1 * 16 * 64 + 32, 64); s1 = WMMA_BF16(aq1, bk, s1);
      }
      // ---- scale + causal/padding mask ----
#pragma unroll
      for (int r = 0; r < 8; ++r) {
        int q   = q0 + moff + r;
        int kc0 = k0 + nloc;
        int kc1 = k0 + 16 + nloc;
        float v0 = s0[r] * 0.125f;   // 1/sqrt(64)
        float v1 = s1[r] * 0.125f;
        v0 = (kc0 > q || kc0 >= len) ? -3.0e38f : v0;
        v1 = (kc1 > q || kc1 >= len) ? -3.0e38f : v1;
        s0[r] = v0; s1[r] = v1;
      }
      // ---- online softmax (rows live in 16-lane halves) ----
#pragma unroll
      for (int r = 0; r < 8; ++r) {
        float mx = fmaxf(s0[r], s1[r]);
#pragma unroll
        for (int off = 8; off >= 1; off >>= 1) mx = fmaxf(mx, __shfl_xor(mx, off, 32));
        float mnew = fmaxf(mrow[r], mx);
        float sf = __expf(mrow[r] - mnew);
        float p0 = __expf(s0[r] - mnew);
        float p1 = __expf(s1[r] - mnew);
        float rs = p0 + p1;
#pragma unroll
        for (int off = 8; off >= 1; off >>= 1) rs += __shfl_xor(rs, off, 32);
        lrow[r] = lrow[r] * sf + rs;
        mrow[r] = mnew;
        s0[r] = p0; s1[r] = p1;
#pragma unroll
        for (int j = 0; j < 4; ++j) accO[j][r] *= sf;
      }
      // ---- P: C-layout regs -> LDS (per-wave region) -> A-layout regs ----
      bf16_t* ps = pstage[wave];
#pragma unroll
      for (int r = 0; r < 8; ++r) {
        int m = moff + r;
        ps[m * 32 + nloc]      = (bf16_t)s0[r];
        ps[m * 32 + 16 + nloc] = (bf16_t)s1[r];
      }
      asm volatile("s_wait_dscnt 0" ::: "memory");
      v16bf ap;
      {
        int m  = lane & 15;
        int kb = (lane >> 4) * 8;
        V16U u;
        u.h[0] = *(const v8bf*)(ps + m * 32 + kb);
        u.h[1] = *(const v8bf*)(ps + m * 32 + kb + 16);
        ap = u.v;
      }
      // ---- O += P @ V  (V in LDS, d-major: contiguous B-operand loads) ----
#pragma unroll
      for (int j = 0; j < 4; ++j) {
        v16bf bv = load_bt(ldsV + j * 16 * 32, 32);
        accO[j] = WMMA_BF16(ap, bv, accO[j]);
      }
    }
  }

  // ---- normalize + store into (B,T,H) bf16 ----
  bf16_t* op = Ob + ((size_t)(b * T) + q0) * 1024 + h * 64;
#pragma unroll
  for (int j = 0; j < 4; ++j) {
#pragma unroll
    for (int r = 0; r < 8; ++r) {
      int m = moff + r;
      float inv = 1.0f / lrow[r];
      op[(size_t)m * 1024 + j * 16 + nloc] = (bf16_t)(accO[j][r] * inv);
    }
  }
}

// ---------------------------------------------------------------------------
// Kernel 3: output projection.  Y(8192,1024) f32 = Ob(8192,1024) x Wout + b.
// 64x64 wave tiles, same structure as kernel 1.
// ---------------------------------------------------------------------------
__global__ void __launch_bounds__(256)
out_gemm_kernel(const bf16_t* __restrict__ Ob, const bf16_t* __restrict__ WT,
                const float* __restrict__ bias, float* __restrict__ Y) {
  const int K = 1024, NT = 16;  // 1024/64 n-tiles
  int wg = blockIdx.x * 8 + (threadIdx.x >> 5);
  int mt = wg / NT, nt = wg % NT;
  int m0 = mt * 64, n0 = nt * 64;
  int lane = threadIdx.x & 31;

  const v8f vz = {0.f, 0.f, 0.f, 0.f, 0.f, 0.f, 0.f, 0.f};
  v8f acc[4][4];
#pragma unroll
  for (int i = 0; i < 4; ++i)
#pragma unroll
    for (int j = 0; j < 4; ++j) acc[i][j] = vz;

  for (int k0 = 0; k0 < K; k0 += 32) {
    v16bf a[4];
#pragma unroll
    for (int i = 0; i < 4; ++i)
      a[i] = load_a(Ob + (size_t)(m0 + i * 16) * K + k0, K);
#pragma unroll
    for (int j = 0; j < 4; ++j) {
      v16bf b = load_bt(WT + (size_t)(n0 + j * 16) * K + k0, K);
#pragma unroll
      for (int i = 0; i < 4; ++i) acc[i][j] = WMMA_BF16(a[i], b, acc[i][j]);
    }
  }

  int nloc = lane & 15;
  int moff = (lane >> 4) * 8;
#pragma unroll
  for (int i = 0; i < 4; ++i) {
#pragma unroll
    for (int j = 0; j < 4; ++j) {
      int col = n0 + j * 16 + nloc;
      float bv = bias[col];
      int mbase = m0 + i * 16 + moff;
#pragma unroll
      for (int r = 0; r < 8; ++r)
        Y[(size_t)(mbase + r) * 1024 + col] = acc[i][j][r] + bv;
    }
  }
}

// ---------------------------------------------------------------------------
// Host launcher
// ---------------------------------------------------------------------------
extern "C" void kernel_launch(void* const* d_in, const int* in_sizes, int n_in,
                              void* d_out, int out_size, void* d_ws, size_t ws_size,
                              hipStream_t stream) {
  const float*         input = (const float*)d_in[0];          // (4,2048,1024)
  const unsigned char* mask  = (const unsigned char*)d_in[1];  // (4,2048) bool
  const float*         w_qkv = (const float*)d_in[2];          // (1024,3072)
  const float*         b_qkv = (const float*)d_in[3];          // (3072,)
  const float*         w_out = (const float*)d_in[4];          // (1024,1024)
  const float*         b_out = (const float*)d_in[5];          // (1024,)
  float*               Y     = (float*)d_out;                  // (4,2048,1024)

  const size_t MB = 1024ull * 1024ull;
  char* w = (char*)d_ws;
  bf16_t* Xb  = (bf16_t*)(w);              // 16 MB (8192x1024)  (reused as Ob)
  bf16_t* WTq = (bf16_t*)(w + 16 * MB);    //  6 MB (3072x1024)
  bf16_t* WTo = (bf16_t*)(w + 22 * MB);    //  2 MB (1024x1024)
  bf16_t* Qb  = (bf16_t*)(w + 24 * MB);    // 16 MB (64,2048,64)
  bf16_t* Kb  = (bf16_t*)(w + 40 * MB);    // 16 MB (64,2048,64)
  bf16_t* VTb = (bf16_t*)(w + 56 * MB);    // 16 MB (64,64,2048)
  int*    lens = (int*)(w + 72 * MB);      // 16 B
  bf16_t* Ob  = Xb;                        // X is dead after QKV GEMM

  const int nX = 8192 * 1024;
  cvt_bf16_kernel<<<(nX + 255) / 256, 256, 0, stream>>>(input, Xb, nX);
  cvt_transpose_kernel<<<(1024 * 3072 + 255) / 256, 256, 0, stream>>>(w_qkv, WTq, 1024, 3072);
  cvt_transpose_kernel<<<(1024 * 1024 + 255) / 256, 256, 0, stream>>>(w_out, WTo, 1024, 1024);
  compute_len_kernel<<<4, 256, 0, stream>>>(mask, lens);

  // 8192/64 m-tiles * 48 n-tiles = 6144 waves / 8 per block
  qkv_gemm_kernel<<<768, 256, 0, stream>>>(Xb, WTq, b_qkv, Qb, Kb, VTb);
  // 64 (b,h) * 16 q-blocks(128 rows) = 1024 blocks
  attn_kernel<<<1024, 256, 0, stream>>>(Qb, Kb, VTb, lens, Ob);
  // 128 m-tiles * 16 n-tiles = 2048 waves / 8 per block
  out_gemm_kernel<<<256, 256, 0, stream>>>(Ob, WTo, b_out, Y);
}
